// LigerFusedLinearGRPOLoss_32255204393543
// MI455X (gfx1250) — matrix operand order repaired and checked by reading, not compiled, not run on hardware
//
#include <hip/hip_runtime.h>
#include <hip/hip_bf16.h>

// Problem constants (from reference setup_inputs)
#define HID   2048          // H
#define VOC   32000         // V
#define NB    4             // B
#define NT    512           // T
#define BT    2048          // B*T rows per model
#define TM    64            // rows per workgroup tile
#define TN    256           // vocab cols per workgroup tile
#define TKK   32            // K step (one bf16 WMMA)
#define NCH   (VOC / TN)    // 125 vocab chunks
#define NRB   (BT / TM)     // 32 row blocks

typedef __attribute__((ext_vector_type(16))) __bf16 v16bf;
typedef __attribute__((ext_vector_type(2)))  __bf16 v2bf;
typedef __attribute__((ext_vector_type(8)))  float  v8f;

union FragU { uint4 q[2]; v16bf v; };

// pack two f32 -> two bf16 (RNE); clang emits hw cvt when available
__device__ __forceinline__ unsigned int pk_bf16(float a, float b) {
  union { v2bf v; unsigned int u; } r;
  r.v = v2bf{(__bf16)a, (__bf16)b};
  return r.u;
}

// ---- CDNA5 async global->LDS copy (ASYNCcnt-tracked DMA; ISA 15.18.3) ------
__device__ __forceinline__ unsigned lds_off(const void* p) {
  // flat LDS address: [63:32]=shared aperture, [31:0]=LDS byte offset
  return (unsigned)(unsigned long long)(uintptr_t)p;
}
__device__ __forceinline__ void async_copy_b128(unsigned loff, const void* g) {
  asm volatile("global_load_async_to_lds_b128 %0, %1, off"
               :: "v"(loff), "v"((unsigned long long)(uintptr_t)g)
               : "memory");
}
#define ASYNC_WAIT(n) asm volatile("s_wait_asynccnt " n ::: "memory")

// ---------------------------------------------------------------------------
// Kernel 0: bulk f32 -> bf16 conversion (done once; removes per-tile VALU work)
// ---------------------------------------------------------------------------
__global__ __launch_bounds__(256)
void convert_bf16(const float* __restrict__ src, unsigned short* __restrict__ dst,
                  int n8) {
  int i = blockIdx.x * 256 + threadIdx.x;
  if (i < n8) {
    const float4* p = (const float4*)(src + (size_t)i * 8);
    float4 f0 = p[0], f1 = p[1];
    uint4 q;
    q.x = pk_bf16(f0.x, f0.y); q.y = pk_bf16(f0.z, f0.w);
    q.z = pk_bf16(f1.x, f1.y); q.w = pk_bf16(f1.z, f1.w);
    *(uint4*)(dst + (size_t)i * 8) = q;
  }
}

// ---------------------------------------------------------------------------
// Kernel 1: tiled bf16 WMMA GEMM (logits tile) + per-row online softmax stats
// grid = (NRB, NCH, 2 models); block = 256 (8 wave32, each 32x64 of C)
// Double-buffered LDS fed by global_load_async_to_lds_b128.
// WPRE: weights pre-converted to bf16 in ws (preferred; halves weight traffic)
// ---------------------------------------------------------------------------
template<bool WPRE>
__global__ __launch_bounds__(256, 2)
void grpo_gemm_lse(const unsigned short* __restrict__ Xb,  // [2][BT][HID] bf16
                   const unsigned short* __restrict__ Wb,  // [2][VOC][HID] bf16 (WPRE)
                   const float* __restrict__ w, const float* __restrict__ rw,
                   const int*   __restrict__ ids,
                   float* __restrict__ part, float* __restrict__ tgt)
{
  const int m    = blockIdx.z;
  const int nch  = blockIdx.y;        // vocab chunk (slow-varying -> L2 reuse)
  const int rb   = blockIdx.x;        // row block
  const int row0 = rb * TM;
  const int col0 = nch * TN;
  const unsigned short* Xm  = Xb + (size_t)m * BT * HID;
  const unsigned short* Wbm = WPRE ? (Wb + (size_t)m * VOC * HID) : (const unsigned short*)0;
  const float*          Wf  = m ? rw : w;

  union SMem {
    struct { unsigned short A[2][TM][TKK + 8]; unsigned short B[2][TN][TKK + 8]; } ab;
    float C[TM][TN + 4];
  };
  __shared__ __align__(16) SMem sm;

  const int t    = threadIdx.x;
  const int lane = t & 31;
  const int wv   = t >> 5;
  const int wm   = wv >> 2;           // 0..1  (M direction, 32 rows each)
  const int wn   = wv & 3;            // 0..3  (N direction, 64 cols each)
  const int l16  = lane & 15;
  const int hi   = lane >> 4;

  v8f zero = {};
  v8f acc[2][4];
#pragma unroll
  for (int i = 0; i < 2; ++i)
#pragma unroll
    for (int j = 0; j < 4; ++j) acc[i][j] = zero;

  // A-stage: 8 bf16 per thread (4 threads/row); B-stage: one row per thread
  const int ar = t >> 2;
  const int ac = (t & 3) << 3;
  const int br = t;
  const unsigned short* XRow  = Xm + (size_t)(row0 + ar) * HID + ac;
  const unsigned short* WbRow = WPRE ? (Wbm + (size_t)(col0 + br) * HID) : (const unsigned short*)0;
  const float*          WfRow = Wf + (size_t)(col0 + br) * HID;

  // stage K-slice `kk` into LDS buffer `buf`
  auto issue = [&](int kk, int buf) {
    async_copy_b128(lds_off(&sm.ab.A[buf][ar][ac]), XRow + kk);
    if (WPRE) {
      unsigned bo = lds_off(&sm.ab.B[buf][br][0]);
#pragma unroll
      for (int r = 0; r < 4; ++r)
        async_copy_b128(bo + r * 16, WbRow + kk + r * 8);
    } else {
      const float4* p = (const float4*)(WfRow + kk);
      uint4* d = (uint4*)&sm.ab.B[buf][br][0];
#pragma unroll
      for (int r = 0; r < 4; ++r) {
        float4 f0 = p[2 * r], f1 = p[2 * r + 1];
        uint4 q;
        q.x = pk_bf16(f0.x, f0.y); q.y = pk_bf16(f0.z, f0.w);
        q.z = pk_bf16(f1.x, f1.y); q.w = pk_bf16(f1.z, f1.w);
        d[r] = q;
      }
    }
  };

  issue(0, 0);
  const int L = HID / TKK;   // 64 K-slices
  for (int i = 0; i < L; ++i) {
    const int buf = i & 1;
    if (i + 1 < L) {
      issue((i + 1) * TKK, buf ^ 1);   // DMA next slice while computing this one
      if (WPRE) { ASYNC_WAIT("0x5"); } else { ASYNC_WAIT("0x1"); }
    } else {
      ASYNC_WAIT("0x0");
    }
    __syncthreads();   // all waves' writes of buf are complete & visible

    // fragments per ISA 16-bit layout:
    // lane group 0 -> K 0..7 then 16..23 ; lane group 1 -> K 8..15 then 24..31
    FragU af[2], bfr[4];
#pragma unroll
    for (int ii = 0; ii < 2; ++ii) {
      const uint4* pa = (const uint4*)&sm.ab.A[buf][wm * 32 + ii * 16 + l16][0];
      af[ii].q[0] = pa[hi];
      af[ii].q[1] = pa[2 + hi];
    }
#pragma unroll
    for (int j = 0; j < 4; ++j) {
      const uint4* pb = (const uint4*)&sm.ab.B[buf][wn * 64 + j * 16 + l16][0];
      bfr[j].q[0] = pb[hi];
      bfr[j].q[1] = pb[2 + hi];
    }
#pragma unroll
    for (int ii = 0; ii < 2; ++ii)
#pragma unroll
      for (int j = 0; j < 4; ++j)
        acc[ii][j] = __builtin_amdgcn_wmma_f32_16x16x32_bf16(
            false, af[ii].v, false, bfr[j].v, (short)0, acc[ii][j], false, false);

    __syncthreads();   // everyone done reading buf; next iter may overwrite it
  }

  // Spill C tile to LDS (union reuse; loop-ending barrier protects readers)
#pragma unroll
  for (int i = 0; i < 2; ++i)
#pragma unroll
    for (int j = 0; j < 4; ++j)
#pragma unroll
      for (int k = 0; k < 8; ++k)
        sm.C[wm * 32 + i * 16 + hi * 8 + k][wn * 64 + j * 16 + l16] = acc[i][j][k];
  __syncthreads();

  // Per-row reduction over 256 cols: 4 threads/row, 64 cols each
  const int row = t >> 2;
  const int sub = t & 3;
  const int rg  = row0 + row;
  const float* cr = &sm.C[row][sub * 64];
  float mx = -INFINITY;
#pragma unroll 8
  for (int k = 0; k < 64; ++k) mx = fmaxf(mx, cr[k]);
  float se = 0.f, smv = 0.f;
#pragma unroll 8
  for (int k = 0; k < 64; ++k) { float v = cr[k]; se += __expf(v - mx); smv += v; }
  // combine the 4 sub-partials (groups are low lane bits -> xor 1,2 stays in group)
#pragma unroll
  for (int off = 1; off < 4; off <<= 1) {
    float omx = __shfl_xor(mx, off, 32);
    float ose = __shfl_xor(se, off, 32);
    float osm = __shfl_xor(smv, off, 32);
    float nm  = fmaxf(mx, omx);
    se  = se * __expf(mx - nm) + ose * __expf(omx - nm);
    mx  = nm;
    smv += osm;
  }
  if (sub == 0) {
    size_t pidx = ((size_t)(m * NCH + nch) * BT + rg) * 3;
    part[pidx + 0] = mx;
    part[pidx + 1] = se;
    part[pidx + 2] = smv;
    int id = ids[rg];
    if (id >= col0 && id < col0 + TN)
      tgt[m * BT + rg] = sm.C[row][id - col0];
  }
}

// ---------------------------------------------------------------------------
// Kernel 2: merge 125 chunk partials per row -> stable logsumexp, per-token logp
// grid = (BT/256, 2)
// ---------------------------------------------------------------------------
__global__ __launch_bounds__(256)
void grpo_lse_reduce(const float* __restrict__ part, const float* __restrict__ tgt,
                     float* __restrict__ logp, float* __restrict__ lsev,
                     float* __restrict__ rsum, float* __restrict__ out)
{
  const int m  = blockIdx.y;
  const int rg = blockIdx.x * 256 + threadIdx.x;
  float M = -INFINITY, S = 0.f, sum = 0.f;
  for (int c = 0; c < NCH; ++c) {
    const float* q = part + ((size_t)(m * NCH + c) * BT + rg) * 3;
    float cm = q[0], cs = q[1];
    sum += q[2];
    float nm = fmaxf(M, cm);
    S = S * __expf(M - nm) + cs * __expf(cm - nm);
    M = nm;
  }
  float l  = M + logf(S);
  float lp = tgt[m * BT + rg] - l;
  logp[m * BT + rg] = lp;
  lsev[m * BT + rg] = l;
  rsum[m * BT + rg] = sum;
  if (m == 0) out[6 + rg] = lp;   // all_logps [B,T]
}

// ---------------------------------------------------------------------------
// Kernel 3: per-chunk (per-batch-row) loss + metrics, averaged over B
// single block of 256 threads
// ---------------------------------------------------------------------------
__global__ __launch_bounds__(256)
void grpo_finalize(const float* __restrict__ logp, const float* __restrict__ lsev,
                   const float* __restrict__ rsum, const float* __restrict__ mask,
                   const float* __restrict__ adv, float* __restrict__ out)
{
  __shared__ float red[256][8];
  const int t = threadIdx.x;
  float tot0 = 0.f, tot1 = 0.f, tot2 = 0.f, tot3 = 0.f, tot4 = 0.f, tot5 = 0.f;

  for (int b = 0; b < NB; ++b) {
    float s[8]; for (int k = 0; k < 8; ++k) s[k] = 0.f;
    const float a = adv[b];
    for (int tok = t; tok < NT; tok += 256) {
      const int rg = b * NT + tok;
      float lp   = logp[rg];
      float rlp  = logp[BT + rg];
      float mk   = mask[rg];
      float diff = rlp - lp;
      float kl   = __expf(diff) - diff - 1.f;
      float c1   = __expf(lp - lp);                 // old_logps = detach(logps)
      float c2   = fminf(fmaxf(c1, 0.8f), 1.2f);    // clip 1±EPSILON
      float l1   = c1 * a, l2 = c2 * a;
      float pt   = -(fminf(l1, l2) - 0.04f * kl);   // BETA = 0.04
      float cl   = (l1 < l2) ? 1.f : 0.f;
      s[0] += pt * mk;  s[1] += kl * mk;  s[2] += cl * mk;  s[3] += mk;
      s[4] += lp;       s[5] += lp * lp;  s[6] += lsev[rg]; s[7] += rsum[rg];
    }
    for (int k = 0; k < 8; ++k) red[t][k] = s[k];
    __syncthreads();
    if (t == 0) {
      float g[8]; for (int k = 0; k < 8; ++k) g[k] = 0.f;
      for (int i = 0; i < 256; ++i)
        for (int k = 0; k < 8; ++k) g[k] += red[i][k];
      float msum = g[3];
      float mean = g[4] / (float)NT;
      float var  = (g[5] - (float)NT * mean * mean) / (float)(NT - 1);
      tot0 += g[0] / msum;                                    // loss
      tot1 += mean;                                           // logps.mean
      tot2 += sqrtf(fmaxf(var, 0.f));                         // logps.std (ddof=1)
      tot3 += g[7] / ((float)NT * (float)VOC) - g[6] / (float)NT;  // lsm.mean
      tot4 += g[1] / msum;                                    // kl
      tot5 += g[2] / msum;                                    // clip_ratio
    }
    __syncthreads();
  }
  if (t == 0) {
    out[0] = tot0 * 0.25f; out[1] = tot1 * 0.25f; out[2] = tot2 * 0.25f;
    out[3] = tot3 * 0.25f; out[4] = tot4 * 0.25f; out[5] = tot5 * 0.25f;
  }
}

extern "C" void kernel_launch(void* const* d_in, const int* in_sizes, int n_in,
                              void* d_out, int out_size, void* d_ws, size_t ws_size,
                              hipStream_t stream) {
  const float* x   = (const float*)d_in[0];   // _input [B,T,H]
  const float* w   = (const float*)d_in[1];   // weight [V,H]
  const int*   ids = (const int*)  d_in[2];   // input_id [B,T]
  const float* mk  = (const float*)d_in[3];   // attention_mask [B,T]
  const float* adv = (const float*)d_in[4];   // advantage [B,1]
  const float* rx  = (const float*)d_in[5];   // ref_input [B,T,H]
  const float* rw  = (const float*)d_in[6];   // ref_weight [V,H]
  float* out = (float*)d_out;                 // 6 scalars + [B,T] logps
  float* ws  = (float*)d_ws;

  const size_t partF = (size_t)2 * NCH * BT * 3;   // 1.536M floats
  const size_t miscF = (size_t)8 * BT;             // tgt+logp+lse+rsum
  float* part = ws;
  float* tgt  = part + partF;
  float* logp = tgt  + 2 * BT;
  float* lsev = logp + 2 * BT;
  float* rsum = lsev + 2 * BT;

  const size_t xElems = (size_t)2 * BT * HID;      // 8.39M bf16
  const size_t wElems = (size_t)2 * VOC * HID;     // 131M bf16
  unsigned short* Xb = (unsigned short*)(ws + partF + miscF);
  unsigned short* Wb = Xb + xElems;

  const size_t needFull = (partF + miscF) * 4 + (xElems + wElems) * 2;  // ~285 MB

  // activations -> bf16 (always; tiny, and stays L2-resident for the GEMM)
  {
    int n8 = (BT * HID) / 8;           // 524288 per model
    convert_bf16<<<n8 / 256, 256, 0, stream>>>(x,  Xb, n8);
    convert_bf16<<<n8 / 256, 256, 0, stream>>>(rx, Xb + (size_t)BT * HID, n8);
  }

  dim3 g1(NRB, NCH, 2);
  if (ws_size >= needFull) {
    // weights -> bf16 once: halves weight HBM traffic, removes in-loop converts
    int n8 = (int)(((size_t)VOC * HID) / 8);   // 8.192M per model
    convert_bf16<<<n8 / 256, 256, 0, stream>>>(w,  Wb, n8);
    convert_bf16<<<n8 / 256, 256, 0, stream>>>(rw, Wb + (size_t)VOC * HID, n8);
    grpo_gemm_lse<true><<<g1, 256, 0, stream>>>(Xb, Wb, w, rw, ids, part, tgt);
  } else {
    grpo_gemm_lse<false><<<g1, 256, 0, stream>>>(Xb, (const unsigned short*)0,
                                                 w, rw, ids, part, tgt);
  }

  dim3 g2(BT / 256, 2);
  grpo_lse_reduce<<<g2, 256, 0, stream>>>(part, tgt, logp, lsev, rsum, out);
  grpo_finalize<<<1, 256, 0, stream>>>(logp, lsev, rsum, mk, adv, out);
}
